// DecoderBlock_87840671137989
// MI455X (gfx1250) — matrix operand order repaired and checked by reading, not compiled, not run on hardware
//
#include <hip/hip_runtime.h>
#include <hip/hip_bf16.h>

typedef __attribute__((ext_vector_type(16))) _Float16 v16h;
typedef __attribute__((ext_vector_type(8)))  _Float16 v8h;
typedef __attribute__((ext_vector_type(8)))  float    v8f;

#define NB    4
#define LSEQ  4096
#define DMOD  128
#define DIN   256
#define BL    (NB*LSEQ)   /* 16384 */

static __device__ __forceinline__ float silu_f(float x) {
    return x / (1.0f + __expf(-x));
}

/* ------------------------------------------------------------------ */
/* prep: f32->f16 weight conversion (+permute/pad) and A = -exp(Alog) */
/* ------------------------------------------------------------------ */
__global__ void k_prep(const float* __restrict__ in_proj_w,
                       const float* __restrict__ up_w,
                       const float* __restrict__ x_proj_w,
                       const float* __restrict__ out_proj_w,
                       const float* __restrict__ convout_w,
                       const float* __restrict__ A_log,
                       _Float16* __restrict__ w_in,    /* 512x128 */
                       _Float16* __restrict__ w_up,    /* 256x128 (n=o*4+i*2+j) */
                       _Float16* __restrict__ w_xp,    /* 64x256 padded */
                       _Float16* __restrict__ w_op,    /* 128x256 */
                       _Float16* __restrict__ w_co,    /* 64x128 */
                       float*    __restrict__ Aneg)    /* 256x16 */
{
    int i = blockIdx.x * blockDim.x + threadIdx.x;
    const int S0 = 512*128;          /* in_proj */
    const int S1 = 256*128;          /* up      */
    const int S2 = 64*256;           /* x_proj padded */
    const int S3 = 128*256;          /* out_proj */
    const int S4 = 64*128;           /* convout */
    const int S5 = 256*16;           /* Aneg */
    if (i < S0) { w_in[i] = (_Float16)in_proj_w[i]; return; }
    i -= S0;
    if (i < S1) {
        int n = i >> 7, c = i & 127;
        int o = n >> 2, ii = (n >> 1) & 1, jj = n & 1;
        /* up_w layout (C_IN=128, C_SKIP=64, 2, 2) */
        w_up[i] = (_Float16)up_w[(((c*64 + o)*2 + ii)*2) + jj];
        return;
    }
    i -= S1;
    if (i < S2) {
        int n = i >> 8, k = i & 255;
        w_xp[i] = (n < 40) ? (_Float16)x_proj_w[n*256 + k] : (_Float16)0.0f;
        return;
    }
    i -= S2;
    if (i < S3) { w_op[i] = (_Float16)out_proj_w[i]; return; }
    i -= S3;
    if (i < S4) { w_co[i] = (_Float16)convout_w[i]; return; }
    i -= S4;
    if (i < S5) { Aneg[i] = -__expf(A_log[i]); }
}

/* transpose x (B,C,H,W) -> xT (B*H*W, C) f16 for the up-conv GEMM */
__global__ void k_transpose_x(const float* __restrict__ x, _Float16* __restrict__ xT)
{
    int i = blockIdx.x * blockDim.x + threadIdx.x;   /* 4096*128 */
    if (i >= 4096*128) return;
    int m = i >> 7, c = i & 127;
    int b = m >> 10, p = m & 1023;
    xT[i] = (_Float16)x[(b*128 + c)*1024 + p];
}

/* ------------------------------------------------------------------ */
/* WMMA GEMM: C[M,N] = A[M,K] * W^T, W torch-layout (N,K) f16.        */
/* One wave -> 16(M) x 64(N) tile; compile-time K, fully unrolled so  */
/* the scheduler overlaps global_load_b128 clauses with WMMAs.        */
/* ------------------------------------------------------------------ */
template<int K>
__global__ __launch_bounds__(32)
void k_gemm_nt4(const _Float16* __restrict__ A, int lda,
                const _Float16* __restrict__ Bw,
                float* __restrict__ C, int ldc)
{
    const int lane = threadIdx.x;
    const int m0 = blockIdx.y * 16;
    const int n0 = blockIdx.x * 64;
    const int row = lane & 15;
    const int hi  = lane >> 4;

    v8f acc0 = {}, acc1 = {}, acc2 = {}, acc3 = {};
    const _Float16* Arow = A + (long)(m0 + row) * lda + hi * 16;
    const _Float16* B0 = Bw + (long)(n0 +  0 + row) * K + hi * 16;
    const _Float16* B1 = Bw + (long)(n0 + 16 + row) * K + hi * 16;
    const _Float16* B2 = Bw + (long)(n0 + 32 + row) * K + hi * 16;
    const _Float16* B3 = Bw + (long)(n0 + 48 + row) * K + hi * 16;

    #pragma unroll
    for (int k0 = 0; k0 < K; k0 += 32) {
        v8h alo = *(const v8h*)(Arow + k0);
        v8h ahi = *(const v8h*)(Arow + k0 + 8);
        v8h b0l = *(const v8h*)(B0 + k0);
        v8h b0h = *(const v8h*)(B0 + k0 + 8);
        v8h b1l = *(const v8h*)(B1 + k0);
        v8h b1h = *(const v8h*)(B1 + k0 + 8);
        v8h b2l = *(const v8h*)(B2 + k0);
        v8h b2h = *(const v8h*)(B2 + k0 + 8);
        v8h b3l = *(const v8h*)(B3 + k0);
        v8h b3h = *(const v8h*)(B3 + k0 + 8);
        v16h a  = __builtin_shufflevector(alo, ahi, 0,1,2,3,4,5,6,7,8,9,10,11,12,13,14,15);
        v16h b0 = __builtin_shufflevector(b0l, b0h, 0,1,2,3,4,5,6,7,8,9,10,11,12,13,14,15);
        v16h b1 = __builtin_shufflevector(b1l, b1h, 0,1,2,3,4,5,6,7,8,9,10,11,12,13,14,15);
        v16h b2 = __builtin_shufflevector(b2l, b2h, 0,1,2,3,4,5,6,7,8,9,10,11,12,13,14,15);
        v16h b3 = __builtin_shufflevector(b3l, b3h, 0,1,2,3,4,5,6,7,8,9,10,11,12,13,14,15);
        acc0 = __builtin_amdgcn_wmma_f32_16x16x32_f16(false, a, false, b0, (short)0, acc0, false, false);
        acc1 = __builtin_amdgcn_wmma_f32_16x16x32_f16(false, a, false, b1, (short)0, acc1, false, false);
        acc2 = __builtin_amdgcn_wmma_f32_16x16x32_f16(false, a, false, b2, (short)0, acc2, false, false);
        acc3 = __builtin_amdgcn_wmma_f32_16x16x32_f16(false, a, false, b3, (short)0, acc3, false, false);
    }

    const int crow = m0 + hi * 8;
    const int cc = n0 + row;
    #pragma unroll
    for (int i = 0; i < 8; ++i) {
        float* cp = C + (long)(crow + i) * ldc + cc;
        cp[0]  = acc0[i];
        cp[16] = acc1[i];
        cp[32] = acc2[i];
        cp[48] = acc3[i];
    }
}

/* up-conv scatter + skip concat -> seq f16 (B, L, 128) */
__global__ void k_build_seq(const float* __restrict__ up,    /* (4096, 256) */
                            const float* __restrict__ up_b,
                            const float* __restrict__ skip,  /* (B,64,64,64) */
                            _Float16* __restrict__ seq)
{
    int i = blockIdx.x * blockDim.x + threadIdx.x;  /* B*L*128 = 2M */
    if (i >= BL*DMOD) return;
    int bl = i >> 7, c = i & 127;
    int b = bl >> 12, l = bl & 4095;
    float v;
    if (c < 64) {
        int r = l >> 6, q = l & 63;
        int h = r >> 1, ii = r & 1, w = q >> 1, jj = q & 1;
        int m = b*1024 + h*32 + w;
        int n = c*4 + ii*2 + jj;
        v = up[m*256 + n] + up_b[c];
    } else {
        v = skip[((b*64) + (c - 64))*4096 + l];
    }
    seq[i] = (_Float16)v;
}

/* depthwise causal conv1d(4) + bias + SiLU on u = xz[...,:256] */
__global__ void k_conv_silu(const float* __restrict__ xz,
                            const float* __restrict__ cw,  /* (256,1,4) */
                            const float* __restrict__ cb,
                            float* __restrict__ ucf, _Float16* __restrict__ uch)
{
    int i = blockIdx.x * blockDim.x + threadIdx.x;  /* B*L*256 = 4M */
    if (i >= BL*DIN) return;
    int bl = i >> 8, d = i & 255;
    int l = bl & 4095;
    float acc = cb[d];
    #pragma unroll
    for (int t = 0; t < 4; ++t) {
        int ls = l - 3 + t;
        if (ls >= 0) acc = fmaf(xz[(long)(bl - (3 - t))*512 + d], cw[d*4 + t], acc);
    }
    float s = silu_f(acc);
    ucf[i] = s;
    uch[i] = (_Float16)s;
}

/* delta = softplus(dt @ dt_proj_w.T + dt_b), dt = dbc[...,:8] */
__global__ void k_delta(const float* __restrict__ dbc,   /* (BL, 64) */
                        const float* __restrict__ dtw,   /* (256, 8) */
                        const float* __restrict__ dtb,
                        float* __restrict__ delta)
{
    int i = blockIdx.x * blockDim.x + threadIdx.x;  /* BL*256 */
    if (i >= BL*DIN) return;
    int bl = i >> 8, d = i & 255;
    const float* row = dbc + (long)bl*64;
    float acc = dtb[d];
    #pragma unroll
    for (int r = 0; r < 8; ++r) acc = fmaf(row[r], dtw[d*8 + r], acc);
    delta[i] = (acc > 20.0f) ? acc : log1pf(__expf(acc));
}

/* selective scan, software-pipelined, gate fused into the store.
   One lane per (d,n) state; only h = fma(dA,h,dBu) is sequential. */
__global__ __launch_bounds__(32)
void k_scan(const float* __restrict__ delta, const float* __restrict__ uc,
            const float* __restrict__ dbc, const float* __restrict__ xz,
            const float* __restrict__ Aneg, const float* __restrict__ Dp,
            _Float16* __restrict__ gated)
{
    int bid = blockIdx.x;                /* 0..511 */
    int b = bid >> 7;
    int d0 = (bid & 127) * 2;
    int lane = threadIdx.x;
    int half = lane >> 4;
    int n = lane & 15;
    int d = d0 + half;

    float Aln = Aneg[d*16 + n];
    float Dpd = Dp[d];
    float h = 0.0f;
    const float* dptr = delta + (long)b*LSEQ*DIN + d;
    const float* uptr = uc    + (long)b*LSEQ*DIN + d;
    const float* bc   = dbc   + (long)b*LSEQ*64;
    const float* zptr = xz    + (long)b*LSEQ*512 + 256 + d;
    _Float16* gp = gated + (long)b*LSEQ*DIN + d;

#define SCAN_STEP(DV, UV, BV, CV, ZV, LI)                                   \
    {                                                                       \
        float dA = __expf((DV) * Aln);                                      \
        h = fmaf(dA, h, (DV) * (UV) * (BV));                                \
        float p = h * (CV);                                                 \
        p += __shfl_xor(p, 1, 16);                                          \
        p += __shfl_xor(p, 2, 16);                                          \
        p += __shfl_xor(p, 4, 16);                                          \
        p += __shfl_xor(p, 8, 16);                                          \
        if (n == 0)                                                         \
            gp[(long)(LI)*DIN] =                                            \
                (_Float16)((p + (UV)*Dpd) * silu_f(ZV));                    \
    }

    float dv = dptr[0];
    float uv = uptr[0];
    float Bv = bc[8 + n];
    float Cv = bc[24 + n];
    float zv = zptr[0];

    for (int l = 0; l < LSEQ - 1; ++l) {
        /* issue step l+1's loads before computing step l */
        float dv2 = dptr[(long)(l + 1)*DIN];
        float uv2 = uptr[(long)(l + 1)*DIN];
        float Bv2 = bc[(l + 1)*64 + 8 + n];
        float Cv2 = bc[(l + 1)*64 + 24 + n];
        float zv2 = zptr[(long)(l + 1)*512];
        SCAN_STEP(dv, uv, Bv, Cv, zv, l);
        dv = dv2; uv = uv2; Bv = Bv2; Cv = Cv2; zv = zv2;
    }
    SCAN_STEP(dv, uv, Bv, Cv, zv, LSEQ - 1);
#undef SCAN_STEP
}

/* wave-per-row LayerNorm over 128 channels + SiLU -> f16 */
__global__ __launch_bounds__(256)
void k_ln_silu(const float* __restrict__ mo, const float* __restrict__ gamma,
               const float* __restrict__ beta, _Float16* __restrict__ out)
{
    int row = blockIdx.x * 8 + (threadIdx.x >> 5);
    if (row >= BL) return;
    int lane = threadIdx.x & 31;
    const float* r = mo + (long)row * DMOD;
    float v[4];
    float s = 0.0f, sq = 0.0f;
    #pragma unroll
    for (int k = 0; k < 4; ++k) {
        v[k] = r[lane + 32*k];
        s += v[k];
        sq += v[k]*v[k];
    }
    #pragma unroll
    for (int m = 1; m < 32; m <<= 1) {
        s  += __shfl_xor(s,  m, 32);
        sq += __shfl_xor(sq, m, 32);
    }
    float mean = s * (1.0f/128.0f);
    float var  = sq * (1.0f/128.0f) - mean*mean;
    float rstd = rsqrtf(var + 1e-5f);
    _Float16* o = out + (long)row * DMOD;
    #pragma unroll
    for (int k = 0; k < 4; ++k) {
        int c = lane + 32*k;
        float xn = (v[k] - mean) * rstd * gamma[c] + beta[c];
        o[c] = (_Float16)silu_f(xn);
    }
}

/* (BL,64) row-major -> (B,64,64,64) NCHW + bias */
__global__ void k_writeout(const float* __restrict__ co, const float* __restrict__ cb,
                           float* __restrict__ out)
{
    int i = blockIdx.x * blockDim.x + threadIdx.x;  /* B*64*4096 */
    if (i >= NB*64*4096) return;
    int b = i >> 18;
    int rem = i & 262143;
    int o = rem >> 12, l = rem & 4095;
    out[i] = co[(long)(b*4096 + l)*64 + o] + cb[o];
}

/* ------------------------------------------------------------------ */
extern "C" void kernel_launch(void* const* d_in, const int* in_sizes, int n_in,
                              void* d_out, int out_size, void* d_ws, size_t ws_size,
                              hipStream_t stream)
{
    const float* x        = (const float*)d_in[0];
    const float* skip_x   = (const float*)d_in[1];
    const float* up_w     = (const float*)d_in[2];
    const float* up_b     = (const float*)d_in[3];
    const float* in_pw    = (const float*)d_in[4];
    const float* conv_w   = (const float*)d_in[5];
    const float* conv_b   = (const float*)d_in[6];
    const float* xp_w     = (const float*)d_in[7];
    const float* dt_w     = (const float*)d_in[8];
    const float* dt_b     = (const float*)d_in[9];
    const float* A_log    = (const float*)d_in[10];
    const float* Dp       = (const float*)d_in[11];
    const float* op_w     = (const float*)d_in[12];
    const float* ln_g     = (const float*)d_in[13];
    const float* ln_b     = (const float*)d_in[14];
    const float* co_w     = (const float*)d_in[15];
    const float* co_b     = (const float*)d_in[16];
    float* out = (float*)d_out;

    char* ws = (char*)d_ws;
    size_t off = 0;
    auto alloc = [&](size_t bytes) { size_t o = off; off += (bytes + 255) & ~size_t(255); return o; };
    float*    Aneg  = (float*)   (ws + alloc(256*16*4));
    _Float16* w_in  = (_Float16*)(ws + alloc(512*128*2));
    _Float16* w_up  = (_Float16*)(ws + alloc(256*128*2));
    _Float16* w_xp  = (_Float16*)(ws + alloc(64*256*2));
    _Float16* w_op  = (_Float16*)(ws + alloc(128*256*2));
    _Float16* w_co  = (_Float16*)(ws + alloc(64*128*2));
    _Float16* xT    = (_Float16*)(ws + alloc((size_t)4096*128*2));
    float*    up    = (float*)   (ws + alloc((size_t)4096*256*4));
    _Float16* seq   = (_Float16*)(ws + alloc((size_t)BL*128*2));
    float*    xz    = (float*)   (ws + alloc((size_t)BL*512*4));
    float*    ucf   = (float*)   (ws + alloc((size_t)BL*256*4));
    _Float16* uch   = (_Float16*)(ws + alloc((size_t)BL*256*2));
    float*    dbc   = (float*)   (ws + alloc((size_t)BL*64*4));
    float*    delta = (float*)   (ws + alloc((size_t)BL*256*4));
    _Float16* gated = (_Float16*)(ws + alloc((size_t)BL*256*2));
    float*    mo    = (float*)   (ws + alloc((size_t)BL*128*4));
    _Float16* lnh   = (_Float16*)(ws + alloc((size_t)BL*128*2));
    float*    co    = (float*)   (ws + alloc((size_t)BL*64*4));
    (void)ws_size; (void)in_sizes; (void)n_in; (void)out_size;

    /* 1. weight prep */
    {
        int n = 512*128 + 256*128 + 64*256 + 128*256 + 64*128 + 256*16;
        k_prep<<<(n + 255)/256, 256, 0, stream>>>(in_pw, up_w, xp_w, op_w, co_w,
                                                  A_log, w_in, w_up, w_xp, w_op, w_co, Aneg);
    }
    /* 2. transpose x for up-conv GEMM */
    k_transpose_x<<<(4096*128 + 255)/256, 256, 0, stream>>>(x, xT);
    /* 3. up-conv GEMM: (4096,128) x (128->256) */
    k_gemm_nt4<128><<<dim3(256/64, 4096/16), 32, 0, stream>>>(xT, 128, w_up, up, 256);
    /* 4. scatter + concat -> seq f16 */
    k_build_seq<<<(BL*DMOD + 255)/256, 256, 0, stream>>>(up, up_b, skip_x, seq);
    /* 5. in_proj GEMM: (16384,128) x (128->512) */
    k_gemm_nt4<128><<<dim3(512/64, BL/16), 32, 0, stream>>>(seq, 128, w_in, xz, 512);
    /* 6. depthwise conv1d + SiLU */
    k_conv_silu<<<(BL*DIN + 255)/256, 256, 0, stream>>>(xz, conv_w, conv_b, ucf, uch);
    /* 7. x_proj GEMM: (16384,256) x (256->64 padded) */
    k_gemm_nt4<256><<<dim3(64/64, BL/16), 32, 0, stream>>>(uch, 256, w_xp, dbc, 64);
    /* 8. dt_proj + softplus */
    k_delta<<<(BL*DIN + 255)/256, 256, 0, stream>>>(dbc, dt_w, dt_b, delta);
    /* 9. selective scan (fused gate): 512 waves, lane = (d,n) state */
    k_scan<<<NB*DIN/2, 32, 0, stream>>>(delta, ucf, dbc, xz, Aneg, Dp, gated);
    /* 10. out_proj GEMM: (16384,256) x (256->128) */
    k_gemm_nt4<256><<<dim3(128/64, BL/16), 32, 0, stream>>>(gated, 256, w_op, mo, 128);
    /* 11. LayerNorm + SiLU */
    k_ln_silu<<<BL/8, 256, 0, stream>>>(mo, ln_g, ln_b, lnh);
    /* 12. 1x1 conv-out GEMM: (16384,128) x (128->64) */
    k_gemm_nt4<128><<<dim3(64/64, BL/16), 32, 0, stream>>>(lnh, 128, w_co, co, 64);
    /* 13. NCHW epilogue + bias */
    k_writeout<<<(NB*64*4096 + 255)/256, 256, 0, stream>>>(co, co_b, out);
}
